// RGCN_4896262718105
// MI455X (gfx1250) — matrix operand order repaired and checked by reading, not compiled, not run on hardware
//
#include <hip/hip_runtime.h>

#define D 128
#define LDSW 136   // LDS row stride in bf16 elements (128 + 8 pad -> conflict-free b128 reads)

typedef __attribute__((ext_vector_type(16))) __bf16 v16bf;
typedef __attribute__((ext_vector_type(8)))  float  v8f;

union Frag {
    v16bf v;
    unsigned short s[16];
    uint4 u4[2];
};

// Native cast -> backend can use v_cvt_pk_bf16_f32 (RNE in hardware)
__device__ __forceinline__ unsigned short f2bf(float f) {
    union { __bf16 b; unsigned short s; } u;
    u.b = (__bf16)f;
    return u.s;
}

// Convert W [R][D][D] (row k, col f) + root [D][D] into bf16, transposed to
// Wb[m][f][k] so each WMMA B-lane reads two 16-byte runs of contiguous k.
__global__ void conv_w_kernel(const float* __restrict__ W, const float* __restrict__ root,
                              unsigned short* __restrict__ Wb, int R) {
    int tid = blockIdx.x * blockDim.x + threadIdx.x;
    int total = (R + 1) * D * D;
    if (tid >= total) return;
    int m = tid / (D * D);
    int rem = tid - m * D * D;
    int f = rem / D;
    int k = rem - f * D;
    float v = (m < R) ? W[(size_t)m * D * D + (size_t)k * D + f] : root[(size_t)k * D + f];
    Wb[tid] = f2bf(v);
}

__global__ void count_kernel(const int* __restrict__ dst, const int* __restrict__ et,
                             int* __restrict__ cnt, int E, int R) {
    int e = blockIdx.x * blockDim.x + threadIdx.x;
    if (e < E) atomicAdd(&cnt[(size_t)dst[e] * R + et[e]], 1);
}

__global__ void coeff_kernel(const int* __restrict__ cnt, float* __restrict__ coeff, int NR) {
    int i = blockIdx.x * blockDim.x + threadIdx.x;
    if (i < NR) {
        int c = cnt[i];
        coeff[i] = (c > 0) ? (1.0f / (float)c) : 0.0f;
    }
}

// h0[n] = emb[x[n]] ; 32 float4 per row
__global__ void gather_kernel(const int* __restrict__ x, const float* __restrict__ emb,
                              float* __restrict__ h, int N) {
    int tid = blockIdx.x * blockDim.x + threadIdx.x;
    if (tid >= N * (D / 4)) return;
    int n = tid >> 5;
    int j = tid & 31;
    const float4* s = reinterpret_cast<const float4*>(emb + (size_t)x[n] * D) + j;
    float4* d = reinterpret_cast<float4*>(h + (size_t)n * D) + j;
    *d = *s;
}

// One wave per edge: agg[et][dst] += coeff * h[src]  (fp32 atomics, full precision)
__global__ void scatter_kernel(const float* __restrict__ h, const float* __restrict__ coeff,
                               const int* __restrict__ src, const int* __restrict__ dst,
                               const int* __restrict__ et, float* __restrict__ agg,
                               int E, int N, int R) {
    int e = blockIdx.x * 8 + (threadIdx.x >> 5);
    if (e >= E) return;
    int lane = threadIdx.x & 31;
    int s = src[e], d = dst[e], r = et[e];
    float c = coeff[(size_t)d * R + r];
    float4 v = *(reinterpret_cast<const float4*>(h + (size_t)s * D) + lane);
    float* p = agg + ((size_t)r * N + d) * D + (size_t)lane * 4;
    atomicAdd(p + 0, v.x * c);
    atomicAdd(p + 1, v.y * c);
    atomicAdd(p + 2, v.z * c);
    atomicAdd(p + 3, v.w * c);
}

// out[16 x 128 tile] = sum_{m=0..R-1} agg_m @ W_m + h @ root  (+bias, opt. ReLU)
// Block = 256 threads = 8 waves; wave w -> output cols [16w, 16w+16).
// A tile (16x128) is staged in LDS as bf16 ONCE per block per source matrix and
// shared by all 8 waves -> 8x fewer conversions and global A reads.
__global__ void __launch_bounds__(256)
rgcn_gemm_kernel(const float* __restrict__ agg, const float* __restrict__ h,
                 const unsigned short* __restrict__ Wb, const float* __restrict__ bias,
                 float* __restrict__ out, int N, int R, int relu) {
    __shared__ unsigned short As[16 * LDSW];

    int tile = blockIdx.x;
    int tid  = threadIdx.x;
    int wave = tid >> 5;
    int lane = tid & 31;
    int hiHalf = lane >> 4;               // 0 for lanes 0-15, 1 for 16-31
    int f0  = wave * 16;
    int col = f0 + (lane & 15);           // output / B column
    int arw = lane & 15;                  // A-fragment row within tile

    // cooperative load/convert mapping: thread -> (row, 8-col segment)
    int crow = tid >> 4;                  // 0..15
    int cseg = (tid & 15) * 8;            // 0,8,...,120

    v8f acc = {};

    for (int m = 0; m <= R; ++m) {
        const float* Atile = ((m < R) ? (agg + (size_t)m * N * D) : h)
                             + (size_t)tile * 16 * D;

        // --- stage 16x128 fp32 -> bf16 into LDS (once, all waves) ---
        const float4* g = reinterpret_cast<const float4*>(Atile + (size_t)crow * D + cseg);
        float4 x0 = g[0];
        float4 x1 = g[1];
        union { unsigned short s[8]; uint4 u; } cv;
        cv.s[0] = f2bf(x0.x); cv.s[1] = f2bf(x0.y); cv.s[2] = f2bf(x0.z); cv.s[3] = f2bf(x0.w);
        cv.s[4] = f2bf(x1.x); cv.s[5] = f2bf(x1.y); cv.s[6] = f2bf(x1.z); cv.s[7] = f2bf(x1.w);
        __syncthreads();   // previous iteration's fragment reads complete (WAR)
        *reinterpret_cast<uint4*>(&As[crow * LDSW + cseg]) = cv.u;
        __syncthreads();   // tile visible to all waves

        const unsigned short* brow = Wb + ((size_t)m * D + col) * D;
        #pragma unroll
        for (int kk = 0; kk < D; kk += 32) {
            // A fragment (16x32 bf16): lanes 0-15 hold K = kk..kk+7, kk+16..kk+23
            //                          lanes 16-31 hold K = kk+8..kk+15, kk+24..kk+31
            int ka = kk + hiHalf * 8;
            Frag a, b;
            a.u4[0] = *reinterpret_cast<const uint4*>(&As[arw * LDSW + ka]);
            a.u4[1] = *reinterpret_cast<const uint4*>(&As[arw * LDSW + ka + 16]);

            // B fragment (32x16 bf16): lanes 0-15 hold K = kk..kk+7, kk+16..kk+23
            //                          lanes 16-31 hold K = kk+8..kk+15, kk+24..kk+31
            int kb0 = kk + hiHalf * 8;
            const uint4* bp = reinterpret_cast<const uint4*>(brow + kb0);
            b.u4[0] = bp[0];
            b.u4[1] = bp[2];

            acc = __builtin_amdgcn_wmma_f32_16x16x32_bf16(false, a.v, false, b.v,
                                                          (short)0, acc, false, false);
        }
    }

    // Epilogue: C/D layout -> element i of lane l is row (i + 8*hiHalf), col (l&15)
    float bv = bias[col];
    int rbase = tile * 16 + hiHalf * 8;
    float* orow = out + (size_t)rbase * D + col;
    #pragma unroll
    for (int i = 0; i < 8; ++i) {
        float v = acc[i] + bv;
        if (relu) v = fmaxf(v, 0.0f);
        orow[(size_t)i * D] = v;
    }
}

extern "C" void kernel_launch(void* const* d_in, const int* in_sizes, int n_in,
                              void* d_out, int out_size, void* d_ws, size_t ws_size,
                              hipStream_t stream) {
    const int*   x     = (const int*)d_in[0];
    const int*   ei    = (const int*)d_in[1];
    const int*   et    = (const int*)d_in[2];
    const float* emb   = (const float*)d_in[3];
    const float* W1    = (const float*)d_in[4];
    const float* root1 = (const float*)d_in[5];
    const float* b1    = (const float*)d_in[6];
    const float* W2    = (const float*)d_in[7];
    const float* root2 = (const float*)d_in[8];
    const float* b2    = (const float*)d_in[9];
    float* out = (float*)d_out;

    const int N = in_sizes[0];
    const int E = in_sizes[2];
    const int R = in_sizes[4] / (D * D);
    const int* src = ei;
    const int* dst = ei + E;

    // Workspace carve-up (~260 MB total)
    char* ws = (char*)d_ws;
    float* coeff = (float*)ws;                  ws += (size_t)N * R * 4;
    int*   cnt   = (int*)ws;                    ws += (size_t)N * R * 4;
    unsigned short* Wb1 = (unsigned short*)ws;  ws += (size_t)(R + 1) * D * D * 2;
    unsigned short* Wb2 = (unsigned short*)ws;  ws += (size_t)(R + 1) * D * D * 2;
    float* h0  = (float*)ws;                    ws += (size_t)N * D * 4;
    float* h1  = (float*)ws;                    ws += (size_t)N * D * 4;
    float* agg = (float*)ws;   // [R][N][D]

    // Weight convert + transpose to bf16 (tiny, L2-resident thereafter)
    int wcnt = (R + 1) * D * D;
    conv_w_kernel<<<(wcnt + 255) / 256, 256, 0, stream>>>(W1, root1, Wb1, R);
    conv_w_kernel<<<(wcnt + 255) / 256, 256, 0, stream>>>(W2, root2, Wb2, R);

    // Per-(dst,rel) mean coefficients — edge structure is shared by both layers
    hipMemsetAsync(cnt, 0, (size_t)N * R * 4, stream);
    count_kernel<<<(E + 255) / 256, 256, 0, stream>>>(dst, et, cnt, E, R);
    coeff_kernel<<<(N * R + 255) / 256, 256, 0, stream>>>(cnt, coeff, N * R);

    // h0 = emb[x]
    gather_kernel<<<(N * 32 + 255) / 256, 256, 0, stream>>>(x, emb, h0, N);

    size_t aggBytes = (size_t)R * N * D * 4;

    // Layer 1: aggregate -> WMMA transform (+bias, ReLU)
    hipMemsetAsync(agg, 0, aggBytes, stream);
    scatter_kernel<<<(E + 7) / 8, 256, 0, stream>>>(h0, coeff, src, dst, et, agg, E, N, R);
    rgcn_gemm_kernel<<<N / 16, 256, 0, stream>>>(agg, h0, Wb1, b1, h1, N, R, 1);

    // Layer 2
    hipMemsetAsync(agg, 0, aggBytes, stream);
    scatter_kernel<<<(E + 7) / 8, 256, 0, stream>>>(h1, coeff, src, dst, et, agg, E, N, R);
    rgcn_gemm_kernel<<<N / 16, 256, 0, stream>>>(agg, h1, Wb2, b2, out, N, R, 0);
}
